// LSTM_50311246905625
// MI455X (gfx1250) — matrix operand order repaired and checked by reading, not compiled, not run on hardware
//
#include <hip/hip_runtime.h>

// LSTM: H=E=1024, T=512, B=128.
// bf16 WMMA (16x16x32, f32 accum). Fused z_t = [x_t | h_t] @ [Wx;Wh]^T per step
// (K=2048); weights pre-transposed/converted to bf16 once (16.8MB, L2-resident
// across the 511 sequential step launches).
// Step kernel: grid (16 col-blocks x 4 row-blocks), block 256 = 8 wave32.
// Block tile = [32 batch x 64 H] cells; wave w owns col-subtile cw=w&3 and gate
// pair g0=(w>>2)*2, computing 4 WMMA tiles (2 row halves x 2 gates) per K-step.

static constexpr int kT  = 512;
static constexpr int kB  = 128;
static constexpr int kE  = 1024;
static constexpr int kH  = 1024;
static constexpr int kK2 = 2048;   // E + H
static constexpr int k4H = 4096;

typedef __attribute__((ext_vector_type(16))) __bf16 v16bf;
typedef __attribute__((ext_vector_type(8)))  float  v8f;

union FragB16 { v16bf v; uint4 q[2]; };

// ---- one-time weight prep: Wt[n][k] = (k<1024 ? Wx[k][n] : Wh[k-1024][n]) as bf16
__global__ __launch_bounds__(256) void lstm_prep_wt(
    const float* __restrict__ Wx, const float* __restrict__ Wh,
    __bf16* __restrict__ Wt)
{
  size_t idx = (size_t)blockIdx.x * 256u + threadIdx.x;  // idx = n*2048 + k
  int    k   = (int)(idx & (kK2 - 1));
  size_t n   = idx >> 11;
  float  v   = (k < kE) ? Wx[(size_t)k * k4H + n]
                        : Wh[(size_t)(k - kE) * k4H + n];
  Wt[idx] = (__bf16)v;
}

// ---- zero h0 (bf16), c (f32), and output slot 0 (h_total[0] == zeros)
__global__ __launch_bounds__(256) void lstm_init(
    float* __restrict__ out0, float* __restrict__ c, __bf16* __restrict__ h0)
{
  int idx = blockIdx.x * 256 + threadIdx.x;   // B*H = 131072 elements
  out0[idx] = 0.0f;
  c[idx]    = 0.0f;
  h0[idx]   = (__bf16)0.0f;
}

__device__ __forceinline__ float sigmoidf_(float x) {
  return 1.0f / (1.0f + __expf(-x));
}

// One K-step: build 2 A-frags (row halves) from LDS, 2 B-frags (gates) from L2,
// issue 4 WMMAs.  Fragment layouts per ISA 7.12.2:
//  A 16x32 bf16: lanes 0-15 row l, K 0-7 & 16-23; lanes 16-31 row l-16, K 8-15 & 24-31.
//  B 32x16 bf16: lane = column; lanes 0-15 K 0-7 & 16-23; lanes 16-31 K 8-15 & 24-31.
__device__ __forceinline__ void mma_step(
    const __bf16* __restrict__ buf, int l16, int half,
    const __bf16* __restrict__ p0, const __bf16* __restrict__ p1,
    v8f& a00, v8f& a01, v8f& a10, v8f& a11)
{
  FragB16 fa0, fa1, fb0, fb1;
  const __bf16* ap0 = buf + l16 * 32 + half * 8;   // rows 0..15 of block tile
  const __bf16* ap1 = ap0 + 16 * 32;               // rows 16..31
  fa0.q[0] = *(const uint4*)(ap0);
  fa0.q[1] = *(const uint4*)(ap0 + 16);
  fa1.q[0] = *(const uint4*)(ap1);
  fa1.q[1] = *(const uint4*)(ap1 + 16);
  fb0.q[0] = *(const uint4*)(p0);
  fb0.q[1] = *(const uint4*)(p0 + 16);
  fb1.q[0] = *(const uint4*)(p1);
  fb1.q[1] = *(const uint4*)(p1 + 16);
  a00 = __builtin_amdgcn_wmma_f32_16x16x32_bf16(false, fa0.v, false, fb0.v,
                                                (short)0, a00, false, false);
  a01 = __builtin_amdgcn_wmma_f32_16x16x32_bf16(false, fa0.v, false, fb1.v,
                                                (short)0, a01, false, false);
  a10 = __builtin_amdgcn_wmma_f32_16x16x32_bf16(false, fa1.v, false, fb0.v,
                                                (short)0, a10, false, false);
  a11 = __builtin_amdgcn_wmma_f32_16x16x32_bf16(false, fa1.v, false, fb1.v,
                                                (short)0, a11, false, false);
}

__global__ __launch_bounds__(256) void lstm_step(
    const float*  __restrict__ xt,    // [128,1024] embeds at step t (f32)
    const __bf16* __restrict__ Wt,    // [4096,2048] transposed bf16 weights
    const float*  __restrict__ bias,  // [4096]
    const __bf16* __restrict__ hin,   // [128,1024] bf16
    __bf16*       __restrict__ hout,
    float*        __restrict__ cbuf,  // [128,1024] f32 cell state
    float*        __restrict__ outt)  // [128,1024] f32 -> h_total[t+1]
{
  __shared__ __align__(16) __bf16 sA[2][32 * 32];  // double-buffered A tile
  __shared__ float sZ[4][32 * 64];                 // activated gates g,i,f,o

  const int tid  = threadIdx.x;
  const int lane = tid & 31;
  const int wave = tid >> 5;
  const int cb   = blockIdx.x;            // 0..15 (64-wide H blocks)
  const int rb   = blockIdx.y;            // 0..3  (32-row batch blocks)
  const int cw   = wave & 3;
  const int g0   = (wave >> 2) * 2;       // 0 -> {g,i}, 2 -> {f,o}
  const int half = lane >> 4;
  const int l16  = lane & 15;
  const int col  = cb * 64 + cw * 16 + l16;   // column within H for this lane

  const __bf16* wb0 = Wt + (size_t)(g0 * kH + col) * kK2;
  const __bf16* wb1 = Wt + (size_t)((g0 + 1) * kH + col) * kK2;

  v8f a00 = {}, a01 = {}, a10 = {}, a11 = {};

  // staging map: thread fills 4 consecutive bf16 (one 8B LDS store)
  const int    srow      = tid >> 3;              // 0..31
  const int    skk       = (tid & 7) * 4;         // 0,4,..,28
  const size_t stage_row = (size_t)(rb * 32 + srow) * 1024 + skk;

  // ---- K half 1: x_t (f32 -> bf16 on the fly) ----
  for (int ki = 0; ki < 32; ++ki) {
    const int k0  = ki * 32;
    __bf16*   dst = sA[ki & 1];
    {
      union { __bf16 h[4]; uint2 u; } pk;
      const float4 v = *(const float4*)(xt + stage_row + k0);
      pk.h[0] = (__bf16)v.x; pk.h[1] = (__bf16)v.y;
      pk.h[2] = (__bf16)v.z; pk.h[3] = (__bf16)v.w;
      *(uint2*)(dst + tid * 4) = pk.u;
    }
    __syncthreads();   // one barrier/iter: double buffering covers WAR reuse
    const __bf16* p0 = wb0 + k0 + half * 8;
    const __bf16* p1 = wb1 + k0 + half * 8;
    __builtin_prefetch(p0 + 32, 0, 1);
    __builtin_prefetch(p1 + 32, 0, 1);
    mma_step(dst, l16, half, p0, p1, a00, a01, a10, a11);
  }

  // ---- K half 2: h_in (already bf16) ----
  for (int ki = 32; ki < 64; ++ki) {
    const int k0  = ki * 32;
    __bf16*   dst = sA[ki & 1];
    *(uint2*)(dst + tid * 4) = *(const uint2*)(hin + stage_row + (k0 - kE));
    __syncthreads();
    const __bf16* p0 = wb0 + k0 + half * 8;
    const __bf16* p1 = wb1 + k0 + half * 8;
    __builtin_prefetch(p0 + 32, 0, 1);
    __builtin_prefetch(p1 + 32, 0, 1);
    mma_step(dst, l16, half, p0, p1, a00, a01, a10, a11);
  }

  // bias + per-gate activation, scatter into LDS gate planes
  const float bv0 = bias[g0 * kH + col];
  const float bv1 = bias[(g0 + 1) * kH + col];
#pragma unroll
  for (int r = 0; r < 8; ++r) {
    const int m  = r + half * 8;              // C/D layout: vgpr r + lane half
    const int nn = cw * 16 + l16;
    const float z00 = a00[r] + bv0, z01 = a01[r] + bv1;
    const float z10 = a10[r] + bv0, z11 = a11[r] + bv1;
    const float v00 = (g0 == 0) ? tanhf(z00) : sigmoidf_(z00);  // gate 0 = tanh
    const float v10 = (g0 == 0) ? tanhf(z10) : sigmoidf_(z10);
    sZ[g0 + 0][m * 64 + nn]        = v00;
    sZ[g0 + 1][m * 64 + nn]        = sigmoidf_(z01);
    sZ[g0 + 0][(m + 16) * 64 + nn] = v10;
    sZ[g0 + 1][(m + 16) * 64 + nn] = sigmoidf_(z11);
  }
  __syncthreads();

  // cross-gate elementwise: c = g*i + c*f ; h = tanh(c)*o   (2048 cells / block)
#pragma unroll
  for (int j = 0; j < 8; ++j) {
    const int    e   = tid + j * 256;
    const int    m   = e >> 6;
    const int    n   = e & 63;
    const size_t idx = (size_t)(rb * 32 + m) * 1024 + (size_t)(cb * 64 + n);
    const float  gg  = sZ[0][e];
    const float  ii  = sZ[1][e];
    const float  ff  = sZ[2][e];
    const float  oo  = sZ[3][e];
    const float  cn  = gg * ii + cbuf[idx] * ff;
    const float  hn  = tanhf(cn) * oo;
    cbuf[idx] = cn;
    hout[idx] = (__bf16)hn;   // next step's A operand
    outt[idx] = hn;           // h_total[t+1] (f32)
  }
}

extern "C" void kernel_launch(void* const* d_in, const int* in_sizes, int n_in,
                              void* d_out, int out_size, void* d_ws, size_t ws_size,
                              hipStream_t stream) {
  const float* embeds = (const float*)d_in[0];  // [T,B,E] f32
  const float* Wx     = (const float*)d_in[1];  // [E,4H]  f32
  const float* Wh     = (const float*)d_in[2];  // [H,4H]  f32
  const float* bias   = (const float*)d_in[3];  // [4H]    f32
  float*       out    = (float*)d_out;          // [T,B,H] f32

  char*   ws  = (char*)d_ws;
  __bf16* Wt  = (__bf16*)ws;                              // 16 MiB
  size_t  off = (size_t)k4H * kK2 * sizeof(__bf16);
  __bf16* h0  = (__bf16*)(ws + off); off += (size_t)kB * kH * sizeof(__bf16);
  __bf16* h1  = (__bf16*)(ws + off); off += (size_t)kB * kH * sizeof(__bf16);
  float*  cb  = (float*)(ws + off);                       // 512 KiB

  lstm_prep_wt<<<(k4H * kK2) / 256, 256, 0, stream>>>(Wx, Wh, Wt);
  lstm_init<<<(kB * kH) / 256, 256, 0, stream>>>(out, cb, h0);

  dim3 grid(16, 4);
  for (int t = 0; t < kT - 1; ++t) {     // h_total[t+1] = h after step t
    const __bf16* hin  = (t & 1) ? h1 : h0;
    __bf16*       hout = (t & 1) ? h0 : h1;
    lstm_step<<<grid, 256, 0, stream>>>(
        embeds + (size_t)t * kB * kE, Wt, bias, hin, hout, cb,
        out + (size_t)(t + 1) * kB * kH);
  }
}
